// AttentionOperation_1039382085848
// MI455X (gfx1250) — matrix operand order repaired and checked by reading, not compiled, not run on hardware
//
#include <hip/hip_runtime.h>
#include <cstddef>

// ---------------------------------------------------------------------------
// CDNA5 / gfx1250 types & WMMA helpers (wave32, v_wmma_f32_16x16x32_bf16)
// ---------------------------------------------------------------------------
typedef __attribute__((ext_vector_type(16))) __bf16 v16bf;
typedef __attribute__((ext_vector_type(8)))  __bf16 v8bf;
typedef __attribute__((ext_vector_type(4)))  __bf16 v4bf;
typedef __attribute__((ext_vector_type(8)))  float  v8f;
typedef __attribute__((ext_vector_type(4)))  float  v4f;

#define DEV __device__ __forceinline__

// native f32 -> bf16 convert (v_cvt_*_bf16_f32 on gfx1250)
DEV __bf16 f2bf(float f) { return (__bf16)f; }

DEV float wave_sum(float v) {
#pragma unroll
  for (int off = 16; off > 0; off >>= 1) v += __shfl_xor(v, off, 32);
  return v;
}
DEV float wave_max(float v) {
#pragma unroll
  for (int off = 16; off > 0; off >>= 1) v = fmaxf(v, __shfl_xor(v, off, 32));
  return v;
}

DEV v8f wmma_bf16(v16bf a, v16bf b, v8f c) {
  // D = A(16x32) * B(32x16) + C(16x16), f32 accumulate
  return __builtin_amdgcn_wmma_f32_16x16x32_bf16(false, a, false, b, (short)0, c,
                                                 false, false);
}

// Fragment gathers per CDNA5 ISA VGPR layouts.
// A (16-bit, 16x32): element e -> K = half*8 + (e&7) + (e&8?16:0), M = lane&15
// Row-major A storage => two contiguous 16B runs per lane (ds_load_b128).
DEV v16bf loadA_row(const __bf16* base, int pitch, int m0, int k0, int lane) {
  int l15 = lane & 15, half = lane >> 4;
  const __bf16* p = base + (size_t)(m0 + l15) * pitch + k0 + half * 8;
  v16bf a;
#pragma unroll
  for (int e = 0; e < 16; ++e) a[e] = p[(e & 7) + ((e & 8) ? 16 : 0)];
  return a;
}
// B (16-bit, 32x16): element e -> K = half*8 + (e&7) + (e&8?16:0), N = lane&15
// Column-major B storage ([n][k]) => two contiguous 16B runs per lane.
DEV v16bf loadB_col(const __bf16* base, int pitch, int k0, int n0, int lane) {
  int l15 = lane & 15, half = lane >> 4;
  const __bf16* p = base + (size_t)(n0 + l15) * pitch + k0 + half * 8;
  v16bf b;
#pragma unroll
  for (int e = 0; e < 16; ++e) b[e] = p[(e & 7) + ((e & 8) ? 16 : 0)];
  return b;
}

// ---------------------------------------------------------------------------
// Problem constants
// ---------------------------------------------------------------------------
constexpr int NB = 8, HH = 8, DDIM = 64, LLEN = 1024, MLEN = 1024, CDIM = 64;
constexpr float BN_EPS = 1e-5f;

// workspace float offsets
constexpr int WS_HS = 0;        // head sum      [8]
constexpr int WS_HSS = 8;       // head sumsq    [8]
constexpr int WS_HA = 16;       // head scale    [8]
constexpr int WS_HB = 24;       // head shift    [8]
constexpr int WS_CS = 32;       // channel sum   [512]
constexpr int WS_CSS = 544;     // channel sumsq [512]
constexpr int WS_CSC = 1056;    // channel scale [512]
constexpr int WS_CSH = 1568;    // channel shift [512]
constexpr int WS_ZERO_N = 1056; // zero [0, 1056)

// kernel1 LDS layout
constexpr int PX = 40;                                 // bf16 pitch, 32-wide chunk
constexpr int K1_SX = 0;                               // [64][PX] bf16
constexpr int K1_GQ = 64 * PX * 2;                     // [64][64] f32
constexpr int K1_RQ = K1_GQ + 64 * 64 * 4;             // [64] f32
constexpr int K1_RK = K1_RQ + 64 * 4;                  // [64] f32
constexpr int SMEM1 = K1_RK + 64 * 4;

// kernel2 LDS layout
constexpr int PQ2 = 72;  // sQ  [32 l][64 d] bf16 pitch (A row-major)
constexpr int PKV = 72;  // sKV tiles, bf16 pitch, double buffered
constexpr int KVSZ = 64 * PKV;                         // elems per buffer
constexpr int PO = 65;   // oT [32 l][64 d] f32 pitch (odd -> conflict-free cols)
constexpr int K2_SS = 0;                               // [32][1024] f32   131072 B
constexpr int K2_SW = 32 * 1024 * 4;                   // [32][1024] bf16   65536 B
constexpr int K2_SQ = K2_SW + 32 * 1024 * 2;           // sQ                 4608 B
constexpr int K2_SKV = K2_SQ + 32 * PQ2 * 2;           // sKV x2            18432 B
constexpr int K2_OT = K2_SKV + 2 * KVSZ * 2;           // oT                 8320 B
constexpr int K2_DEN = K2_OT + 32 * PO * 4;            // denom[32]           128 B
constexpr int SMEM2 = K2_DEN + 32 * 4;                 // ~223 KB (< 320 KB/WGP)

// ---------------------------------------------------------------------------
// Kernel 0: zero stat accumulators (reset every launch for graph replay)
// ---------------------------------------------------------------------------
__global__ void k0_zero(float* ws) {
  for (int i = threadIdx.x; i < WS_ZERO_N; i += 256) ws[i] = 0.f;
}

// ---------------------------------------------------------------------------
// Kernel 1: per-head logit statistics WITHOUT materializing the 256MB logits.
//   sum(S)  = sum_b <rowsum(Q_b), rowsum(K_b)>
//   sum(S^2)= sum_b <Q_b Q_b^T, K_b K_b^T>_F      (Gram matrices via WMMA)
// ---------------------------------------------------------------------------
__global__ void k1_logit_stats(const float* __restrict__ Qg,
                               const float* __restrict__ Kg,
                               float* headSum, float* headSumSq) {
  extern __shared__ char smem[];
  __bf16* sX   = (__bf16*)(smem + K1_SX);
  float*  gq   = (float*)(smem + K1_GQ);
  float*  rowq = (float*)(smem + K1_RQ);
  float*  rowk = (float*)(smem + K1_RK);

  const int t = threadIdx.x, lane = t & 31, wave = t >> 5;
  const int l15 = lane & 15, half = lane >> 4;
  const int bh = blockIdx.x, h = bh & (HH - 1);
  const float* Qp = Qg + (size_t)bh * DDIM * LLEN;
  const float* Kp = Kg + (size_t)bh * DDIM * MLEN;

  // ---- row sums (one wave per 8 rows, float4 loads) ----
  for (int dd = 0; dd < 8; ++dd) {
    int d = wave * 8 + dd;
    const v4f* qr = (const v4f*)(Qp + (size_t)d * LLEN);
    const v4f* kr = (const v4f*)(Kp + (size_t)d * MLEN);
    float sq = 0.f, sk = 0.f;
#pragma unroll
    for (int i = 0; i < 8; ++i) {
      v4f a = qr[lane + 32 * i], b = kr[lane + 32 * i];
      sq += a[0] + a[1] + a[2] + a[3];
      sk += b[0] + b[1] + b[2] + b[3];
    }
    sq = wave_sum(sq);
    sk = wave_sum(sk);
    if (lane == 0) { rowq[d] = sq; rowk[d] = sk; }
  }
  __syncthreads();
  if (t < 64) {
    float p = rowq[t] * rowk[t];
    p = wave_sum(p);
    if (lane == 0) atomicAdd(headSum + h, p);
  }

  // ---- GQ = Q Q^T (64x64, contract over L in 32-wide bf16 chunks) ----
  v8f cq[2] = {};
  for (int lc = 0; lc < 32; ++lc) {
    __syncthreads();
    { int d = t >> 2, j0 = (t & 3) * 8;
      const v4f* src = (const v4f*)(Qp + (size_t)d * LLEN + lc * 32 + j0);
      v4f x0 = src[0], x1 = src[1];
      v8bf o;
#pragma unroll
      for (int i = 0; i < 4; ++i) { o[i] = f2bf(x0[i]); o[4 + i] = f2bf(x1[i]); }
      *(v8bf*)(sX + d * PX + j0) = o; }
    __syncthreads();
#pragma unroll
    for (int ti = 0; ti < 2; ++ti) {
      int tile = wave * 2 + ti, tm = tile >> 2, tn = tile & 3;
      v16bf a = loadA_row(sX, PX, tm * 16, 0, lane);   // A[m=d][k=l]
      v16bf b = loadB_col(sX, PX, 0, tn * 16, lane);   // B[k=l][n=d']
      cq[ti] = wmma_bf16(a, b, cq[ti]);
    }
  }
#pragma unroll
  for (int ti = 0; ti < 2; ++ti) {
    int tile = wave * 2 + ti, tm = tile >> 2, tn = tile & 3;
#pragma unroll
    for (int r = 0; r < 8; ++r)
      gq[(tm * 16 + half * 8 + r) * 64 + tn * 16 + l15] = cq[ti][r];
  }
  __syncthreads();

  // ---- GK = K K^T, fused Frobenius dot with GQ ----
  v8f ck[2] = {};
  for (int lc = 0; lc < 32; ++lc) {
    __syncthreads();
    { int d = t >> 2, j0 = (t & 3) * 8;
      const v4f* src = (const v4f*)(Kp + (size_t)d * MLEN + lc * 32 + j0);
      v4f x0 = src[0], x1 = src[1];
      v8bf o;
#pragma unroll
      for (int i = 0; i < 4; ++i) { o[i] = f2bf(x0[i]); o[4 + i] = f2bf(x1[i]); }
      *(v8bf*)(sX + d * PX + j0) = o; }
    __syncthreads();
#pragma unroll
    for (int ti = 0; ti < 2; ++ti) {
      int tile = wave * 2 + ti, tm = tile >> 2, tn = tile & 3;
      v16bf a = loadA_row(sX, PX, tm * 16, 0, lane);
      v16bf b = loadB_col(sX, PX, 0, tn * 16, lane);
      ck[ti] = wmma_bf16(a, b, ck[ti]);
    }
  }
  float acc = 0.f;
#pragma unroll
  for (int ti = 0; ti < 2; ++ti) {
    int tile = wave * 2 + ti, tm = tile >> 2, tn = tile & 3;
#pragma unroll
    for (int r = 0; r < 8; ++r)
      acc += gq[(tm * 16 + half * 8 + r) * 64 + tn * 16 + l15] * ck[ti][r];
  }
  acc = wave_sum(acc);
  if (lane == 0) atomicAdd(headSumSq + h, acc);
}

// ---------------------------------------------------------------------------
// Kernel 1b: finalize per-head BN affine: s' = s*A_h + B_h
// ---------------------------------------------------------------------------
__global__ void k1b_head_finalize(const float* hS, const float* hSS,
                                  const float* gs, const float* bs,
                                  float* hA, float* hB) {
  int h = threadIdx.x;
  if (h < HH) {
    const float inv = 1.f / (float)((size_t)NB * LLEN * MLEN);
    float mu = hS[h] * inv;
    float var = hSS[h] * inv - mu * mu;
    float A = rsqrtf(var + BN_EPS) * gs[h];
    hA[h] = A;
    hB[h] = bs[h] - mu * A;
  }
}

// ---------------------------------------------------------------------------
// Kernel 2: fused  S = (Q^T K)*A_h+B_h  ->  softmax  ->  O = P V^T
// One workgroup per (b,h, 32-row L tile). 256 threads = 8 waves.
// S (32x1024 f32) + P (bf16) live in LDS; logits never touch HBM.
// Double-buffered K/V staging: one barrier per m-block.
// ---------------------------------------------------------------------------
__global__ void k2_attention(const float* __restrict__ Qg,
                             const float* __restrict__ Kg,
                             const float* __restrict__ Vg,
                             const float* __restrict__ hA,
                             const float* __restrict__ hB,
                             float* __restrict__ rv,
                             float* __restrict__ chSum,
                             float* __restrict__ chSumSq) {
  extern __shared__ char smem[];
  float*  sS  = (float*)(smem + K2_SS);   // [32][1024] f32 logits
  __bf16* sW  = (__bf16*)(smem + K2_SW);  // [32][1024] bf16 unnormalized P
  __bf16* sQ  = (__bf16*)(smem + K2_SQ);  // [32 l][64 d] (A row-major)
  __bf16* sKV = (__bf16*)(smem + K2_SKV); // 2 x [64][PKV]
  float*  oT  = (float*)(smem + K2_OT);   // [32 l][64 d]
  float*  den = (float*)(smem + K2_DEN);  // [32]

  const int t = threadIdx.x, lane = t & 31, wave = t >> 5;
  const int l15 = lane & 15, half = lane >> 4;
  const int wg = blockIdx.x;
  const int lt32 = wg & 31, bh = wg >> 5, h = bh & (HH - 1);
  const int l0 = lt32 * 32;
  const float* Qp = Qg + (size_t)bh * DDIM * LLEN;
  const float* Kp = Kg + (size_t)bh * DDIM * MLEN;
  const float* Vp = Vg + (size_t)bh * CDIM * MLEN;
  const float Ah = hA[h], Bh = hB[h];

  // stage Q tile transposed -> sQ[l][d]  (fp32 -> bf16)
  { int d = t >> 2, j0 = (t & 3) * 8;
    const float* src = Qp + (size_t)d * LLEN + l0 + j0;
#pragma unroll
    for (int i = 0; i < 8; ++i) sQ[(j0 + i) * PQ2 + d] = f2bf(src[i]); }
  __syncthreads();

  // ---- Phase A: S = (Q^T K) * Ah + Bh ----
  const int qlt = wave & 1, qmt = wave >> 1;          // 2 x 4 tiles of 16x16
  // Q A-fragments are m-block invariant: load once, keep in VGPRs
  v16bf aQ0 = loadA_row(sQ, PQ2, qlt * 16, 0, lane);
  v16bf aQ1 = loadA_row(sQ, PQ2, qlt * 16, 32, lane);
  for (int mb = 0; mb < 16; ++mb) {
    __bf16* buf = sKV + (mb & 1) * KVSZ;
    // stage K block TRANSPOSED -> buf[m][d] so B-fragments are contiguous
    { int d = t >> 2, j0 = (t & 3) * 16;
      const v4f* src = (const v4f*)(Kp + (size_t)d * MLEN + mb * 64 + j0);
      v4f x0 = src[0], x1 = src[1], x2 = src[2], x3 = src[3];
#pragma unroll
      for (int i = 0; i < 4; ++i) {
        buf[(j0 + i) * PKV + d]      = f2bf(x0[i]);
        buf[(j0 + 4 + i) * PKV + d]  = f2bf(x1[i]);
        buf[(j0 + 8 + i) * PKV + d]  = f2bf(x2[i]);
        buf[(j0 + 12 + i) * PKV + d] = f2bf(x3[i]);
      } }
    __syncthreads();
    v8f c = {};
    c = wmma_bf16(aQ0, loadB_col(buf, PKV, 0, qmt * 16, lane), c);   // k=d 0..31
    c = wmma_bf16(aQ1, loadB_col(buf, PKV, 32, qmt * 16, lane), c);  // k=d 32..63
#pragma unroll
    for (int r = 0; r < 8; ++r) {
      int row = qlt * 16 + half * 8 + r;
      sS[row * 1024 + mb * 64 + qmt * 16 + l15] = c[r] * Ah + Bh;
    }
  }
  __syncthreads();

  // ---- Phase B: fp32 softmax over M=1024 per row (denominator deferred) ----
  for (int rr = 0; rr < 4; ++rr) {
    int row = wave * 4 + rr;
    const v4f* rp = (const v4f*)(sS + row * 1024);
    v4f vr[8];
    float mx = -3.0e38f;
#pragma unroll
    for (int i = 0; i < 8; ++i) {
      vr[i] = rp[lane + 32 * i];
      mx = fmaxf(mx, fmaxf(fmaxf(vr[i][0], vr[i][1]), fmaxf(vr[i][2], vr[i][3])));
    }
    mx = wave_max(mx);
    float sum = 0.f;
    v4bf* wp = (v4bf*)(sW + row * 1024);
#pragma unroll
    for (int i = 0; i < 8; ++i) {
      float p0 = __expf(vr[i][0] - mx), p1 = __expf(vr[i][1] - mx);
      float p2 = __expf(vr[i][2] - mx), p3 = __expf(vr[i][3] - mx);
      sum += (p0 + p1) + (p2 + p3);
      v4bf pk;
      pk[0] = f2bf(p0); pk[1] = f2bf(p1); pk[2] = f2bf(p2); pk[3] = f2bf(p3);
      wp[lane + 32 * i] = pk;
    }
    sum = wave_sum(sum);
    if (lane == 0) den[row] = sum;
  }
  __syncthreads();

  // ---- Phase C: O = P V^T  (accumulate across M in registers) ----
  const int olt = wave >> 2, odt = wave & 3;          // 2 x 4 tiles of 16x16
  v8f o = {};
  for (int mb = 0; mb < 16; ++mb) {
    __bf16* buf = sKV + (mb & 1) * KVSZ;
    // stage V block natural -> buf[d][m]  (already col-major for B[k=m][n=d])
    { int d = t >> 2, j0 = (t & 3) * 16;
      const v4f* src = (const v4f*)(Vp + (size_t)d * MLEN + mb * 64 + j0);
      v4f x0 = src[0], x1 = src[1], x2 = src[2], x3 = src[3];
      v8bf o0, o1;
#pragma unroll
      for (int i = 0; i < 4; ++i) {
        o0[i] = f2bf(x0[i]); o0[4 + i] = f2bf(x1[i]);
        o1[i] = f2bf(x2[i]); o1[4 + i] = f2bf(x3[i]);
      }
      *(v8bf*)(buf + d * PKV + j0) = o0;
      *(v8bf*)(buf + d * PKV + j0 + 8) = o1; }
    __syncthreads();
#pragma unroll
    for (int ks = 0; ks < 64; ks += 32) {
      v16bf a = loadA_row(sW, 1024, olt * 16, mb * 64 + ks, lane); // A[m=l][k=m]
      v16bf b = loadB_col(buf, PKV, ks, odt * 16, lane);           // B[k=m][n=d]
      o = wmma_bf16(a, b, o);
    }
  }
  // normalize by softmax denom, transpose through LDS
#pragma unroll
  for (int r = 0; r < 8; ++r) {
    int row = olt * 16 + half * 8 + r;
    oT[row * PO + odt * 16 + l15] = o[r] / den[row];
  }
  __syncthreads();

  // ---- write rv (coalesced along L) + per-channel BN partial sums ----
  for (int pass = 0; pass < 8; ++pass) {
    int d = pass * 8 + wave;
    float v = oT[lane * PO + d];
    rv[(size_t)(bh * CDIM + d) * LLEN + l0 + lane] = v;
    float s1 = wave_sum(v);
    float s2 = wave_sum(v * v);
    if (lane == 0) {
      atomicAdd(chSum + h * CDIM + d, s1);
      atomicAdd(chSumSq + h * CDIM + d, s2);
    }
  }
}

// ---------------------------------------------------------------------------
// Kernel 3a: finalize per-channel BN affine
// ---------------------------------------------------------------------------
__global__ void k3a_chan_finalize(const float* cS, const float* cSS,
                                  const float* gv, const float* bv,
                                  float* cSc, float* cSh) {
  int c = blockIdx.x * blockDim.x + threadIdx.x;
  if (c < HH * CDIM) {
    const float inv = 1.f / (float)(NB * LLEN);
    float mu = cS[c] * inv;
    float var = cSS[c] * inv - mu * mu;
    float sc = rsqrtf(var + BN_EPS) * gv[c];
    cSc[c] = sc;
    cSh[c] = bv[c] - mu * sc;
  }
}

// ---------------------------------------------------------------------------
// Kernel 3b: in-place channel affine + exact (erf) GELU, float4 vectorized
// ---------------------------------------------------------------------------
__global__ void k3b_bn_gelu(float* __restrict__ out,
                            const float* __restrict__ cSc,
                            const float* __restrict__ cSh, int total4) {
  v4f* o4 = (v4f*)out;
  for (int idx = blockIdx.x * blockDim.x + threadIdx.x; idx < total4;
       idx += gridDim.x * blockDim.x) {
    v4f x = o4[idx];
    int ch = (idx >> 8) & (HH * CDIM - 1);  // 256 float4 per L-row
    float sc = cSc[ch], sh = cSh[ch];
    v4f y;
#pragma unroll
    for (int j = 0; j < 4; ++j) {
      float v = x[j] * sc + sh;
      y[j] = 0.5f * v * (1.0f + erff(v * 0.70710678118654752f));
    }
    o4[idx] = y;
  }
}

// ---------------------------------------------------------------------------
// Host launcher
// ---------------------------------------------------------------------------
extern "C" void kernel_launch(void* const* d_in, const int* in_sizes, int n_in,
                              void* d_out, int out_size, void* d_ws, size_t ws_size,
                              hipStream_t stream) {
  (void)in_sizes; (void)n_in; (void)out_size; (void)ws_size;
  const float* Q  = (const float*)d_in[0];
  const float* K  = (const float*)d_in[1];
  const float* V  = (const float*)d_in[2];
  const float* gs = (const float*)d_in[3];
  const float* bs = (const float*)d_in[4];
  const float* gv = (const float*)d_in[5];
  const float* bv = (const float*)d_in[6];
  float* out = (float*)d_out;
  float* ws  = (float*)d_ws;

  float* hS  = ws + WS_HS;
  float* hSS = ws + WS_HSS;
  float* hA  = ws + WS_HA;
  float* hB  = ws + WS_HB;
  float* cS  = ws + WS_CS;
  float* cSS = ws + WS_CSS;
  float* cSc = ws + WS_CSC;
  float* cSh = ws + WS_CSH;

  k0_zero<<<1, 256, 0, stream>>>(ws);
  k1_logit_stats<<<NB * HH, 256, SMEM1, stream>>>(Q, K, hS, hSS);
  k1b_head_finalize<<<1, 32, 0, stream>>>(hS, hSS, gs, bs, hA, hB);
  k2_attention<<<NB * HH * (LLEN / 32), 256, SMEM2, stream>>>(Q, K, V, hA, hB,
                                                              out, cS, cSS);
  k3a_chan_finalize<<<2, 256, 0, stream>>>(cS, cSS, gv, bv, cSc, cSh);
  const int total4 = NB * HH * CDIM * LLEN / 4;
  k3b_bn_gelu<<<4096, 256, 0, stream>>>(out, cSc, cSh, total4);
}
